// SkipInteractionBlock_71365176590871
// MI455X (gfx1250) — matrix operand order repaired and checked by reading, not compiled, not run on hardware
//
#include <hip/hip_runtime.h>

// ---------------------------------------------------------------------------
// MACE-style skip-interaction block for MI455X (gfx1250), wave32 + WMMA bf16.
//
// Edge stage:  every einsum against the generated weights tw=ef@Wgen is a
// bilinear form  z[k=r*32+u] = ef[r]*q[u]  times a FIXED 256x32 matrix
// (reshape of Wgen).  Per-edge scalars ys / yv_i commute out of the u-sum,
// so per 16-edge tile we need 6 GEMMs (16x256x32) sharing 5 A's and 4 B's,
// then 4 small GEMMs (16x32 @ 32x32) for the L1 mixes, then atomic scatter.
// Node stage:  same trick with z[k=u*16+v] = M[u]*attr[v], K=512, vs WS0/WS1.
//
// All GEMMs run on v_wmma_f32_16x16x32_bf16 (fp32 accumulate).  B matrices
// are pre-packed (once per launch) in exact B-fragment order so a wave loads
// one 32-byte contiguous vector per fragment.  All normalization constants
// are folded into the packed weights.  A-fragment elements use native __bf16
// casts so the compiler emits hardware v_cvt_pk_bf16_f32 instead of a
// software RNE bit-twiddle.
// ---------------------------------------------------------------------------

#define NNODES 10000
#define NEDGES 100000
#define EDGE_TILES (NEDGES / 16)
#define NODE_TILES (NNODES / 16)

// (1/sqrt(8)) * (1/sqrt(2*MUL)) == 1/sqrt(512) == SNORM
#define WSCALE 0.04419417382415922f
#define SNORM  0.04419417382415922f
#define LNORM  0.17677669529663687f   // 1/sqrt(32)
#define RSQRT3 0.5773502691896258f

typedef __attribute__((ext_vector_type(16))) __bf16 v16bf;
typedef __attribute__((ext_vector_type(8)))  float  v8f;

// workspace layout (bytes); total ~5.26 MB
#define ACC_OFF   0           // float[NNODES*128]  node accumulator (Ms|Mv)
#define WFRAG_OFF 5120000     // u16[32768]  Wgen  frags  [m4][nt2][ks8][lane32][16]
#define WS0F_OFF  5185536     // u16[16384]  WS0   frags  [nt2][ks16][lane32][16]
#define WS1F_OFF  5218304     // u16[16384]  WS1   frags
#define L1SF_OFF  5251072     // u16[1024]   L1s   frags  [nt2][lane32][16]
#define L1VF_OFF  5253120
#define L2SF_OFF  5255168
#define L2VF_OFF  5257216

// exact-RNE bf16 bits (prep path only; cold)
static __device__ __forceinline__ unsigned short f2bfbits(float f) {
  union { float f; unsigned u; } a; a.f = f;
  unsigned r = (a.u + 0x7FFFu + ((a.u >> 16) & 1u)) >> 16;
  return (unsigned short)r;
}
static __device__ __forceinline__ v8f wmma_bf16(v16bf a, v16bf b, v8f c) {
  // 8 args: (neg_a, A, neg_b, B, c_mod, C, reuse_a, reuse_b)
  return __builtin_amdgcn_wmma_f32_16x16x32_bf16(false, a, false, b, (short)0, c,
                                                 false, false);
}
static __device__ __forceinline__ void atomic_add_f32(float* p, float v) {
  __hip_atomic_fetch_add(p, v, __ATOMIC_RELAXED, __HIP_MEMORY_SCOPE_AGENT);
}

// ---------------------------------------------------------------------------
// Prep: zero node accumulator; pack all B matrices into bf16 fragment order.
// B-fragment order used here (contiguous k, same as the A side; 32x16 per k-step):
//   lane L: N = (L&15) + nt*16 ;  element e(0..15): K = ks*32 + (L&16) + e
// ---------------------------------------------------------------------------
__global__ void prep_kernel(const float* __restrict__ Wgen,
                            const float* __restrict__ L1s, const float* __restrict__ L1v,
                            const float* __restrict__ WS0, const float* __restrict__ WS1,
                            const float* __restrict__ L2s, const float* __restrict__ L2v,
                            unsigned short* __restrict__ wfrag,
                            unsigned short* __restrict__ ws0f,
                            unsigned short* __restrict__ ws1f,
                            unsigned short* __restrict__ l1sf,
                            unsigned short* __restrict__ l1vf,
                            unsigned short* __restrict__ l2sf,
                            unsigned short* __restrict__ l2vf,
                            float* __restrict__ acc) {
  int tid = blockIdx.x * blockDim.x + threadIdx.x;
  int stride = gridDim.x * blockDim.x;

  for (int i = tid; i < NNODES * 128; i += stride) acc[i] = 0.0f;

  // Wgen: W_m[K=r*32+u][w] = Wgen[r*4096 + m*1024 + u*32 + w]
  for (int i = tid; i < 32768; i += stride) {
    int elem = i & 15, lane = (i >> 4) & 31, ks = (i >> 9) & 7;
    int nt = (i >> 12) & 1, m = (i >> 13) & 3;
    int N = (lane & 15) + nt * 16;
    int intra = (lane & 16) + elem;                 // u = K & 31, r = ks
    wfrag[i] = f2bfbits(Wgen[ks * 4096 + m * 1024 + intra * 32 + N] * WSCALE);
  }
  // WS: B[K=u*16+v][w] = WS[K*32 + w]
  for (int i = tid; i < 16384; i += stride) {
    int elem = i & 15, lane = (i >> 4) & 31, ks = (i >> 9) & 15;
    int N = (lane & 15) + ((i >> 13) & 1) * 16;
    int K = ks * 32 + (lane & 16) + elem;
    ws0f[i] = f2bfbits(WS0[K * 32 + N] * SNORM);
    ws1f[i] = f2bfbits(WS1[K * 32 + N] * SNORM);
  }
  // L matrices: B[K=w][n] = L[K*32 + n]
  for (int i = tid; i < 1024; i += stride) {
    int elem = i & 15, lane = (i >> 4) & 31;
    int N = (lane & 15) + ((i >> 9) & 1) * 16;
    int K = (lane & 16) + elem;
    l1sf[i] = f2bfbits(L1s[K * 32 + N] * LNORM);
    l1vf[i] = f2bfbits(L1v[K * 32 + N] * LNORM);
    l2sf[i] = f2bfbits(L2s[K * 32 + N] * LNORM);
    l2vf[i] = f2bfbits(L2v[K * 32 + N] * LNORM);
  }
}

// ---------------------------------------------------------------------------
// Edge kernel: one wave32 per 16-edge tile.
// ---------------------------------------------------------------------------
__global__ __launch_bounds__(32)
void edge_kernel(const float* __restrict__ node_feats,
                 const float* __restrict__ edge_attrs,
                 const float* __restrict__ edge_feats,
                 const int*   __restrict__ edge_index,
                 const unsigned short* __restrict__ wfrag,
                 const unsigned short* __restrict__ l1sf,
                 const unsigned short* __restrict__ l1vf,
                 float* __restrict__ acc) {
  __shared__ float s_q[5][16][32];   // 0:xs  1..3:xv_i  4:dot
  __shared__ float s_ef[16][8];
  __shared__ float s_ys[16];
  __shared__ float s_yv[16][3];
  __shared__ int   s_recv[16];
  __shared__ float s_stage[16][32];

  const int lane = threadIdx.x;
  const int eb = blockIdx.x * 16;

  // per-edge scalars
  if (lane < 16) {
    int eg = eb + lane;
    s_recv[lane] = edge_index[NEDGES + eg];
    float4 ea = *(const float4*)(edge_attrs + eg * 4);
    s_ys[lane] = ea.x;
    s_yv[lane][0] = ea.y; s_yv[lane][1] = ea.z; s_yv[lane][2] = ea.w;
#pragma unroll
    for (int r = 0; r < 8; ++r) s_ef[lane][r] = edge_feats[eg * 8 + r];
  }
  // gathered sender features: 2 lanes per edge, 64 floats each
  {
    int e = lane >> 1, h = lane & 1;
    int snd = edge_index[eb + e];
    const float* src = node_feats + snd * 128 + h * 64;
#pragma unroll
    for (int c = 0; c < 64; ++c) {
      int idx = h * 64 + c;
      float v = src[c];
      if (idx < 32) s_q[0][e][idx] = v;                       // xs
      else { int z = idx - 32; s_q[1 + (z % 3)][e][z / 3] = v; }  // xv[:,i]
    }
  }
  __syncthreads();
  // dot[e][u] = (xv . yv) / sqrt(3)
  for (int k = lane; k < 512; k += 32) {
    int e = k >> 5, u = k & 31;
    s_q[4][e][u] = (s_q[1][e][u] * s_yv[e][0] + s_q[2][e][u] * s_yv[e][1] +
                    s_q[3][e][u] * s_yv[e][2]) * RSQRT3;
  }
  __syncthreads();

  const int hi = (lane >> 4) & 1;
  const int mrow = lane & 15;
  // A-fragment u-offsets (contiguous k = hi*16 + s, same order as packed B) -- constant across k-steps
  int uoff[8];
#pragma unroll
  for (int v = 0; v < 8; ++v) uoff[v] = hi * 16 + 2 * v;

  v8f zero = {0.f, 0.f, 0.f, 0.f, 0.f, 0.f, 0.f, 0.f};
  v8f accm[6][2];   // 0:Ts 1:Tb 2..4:Tc_i 5:Td, x 2 N-tiles
#pragma unroll
  for (int g = 0; g < 6; ++g)
#pragma unroll
    for (int nt = 0; nt < 2; ++nt) accm[g][nt] = zero;

  const v16bf* wf = (const v16bf*)wfrag;
  for (int ks = 0; ks < 8; ++ks) {
    float efv = s_ef[mrow][ks];          // r == ks within one k-step
    v16bf A[5];
#pragma unroll
    for (int m = 0; m < 5; ++m)
#pragma unroll
      for (int v = 0; v < 8; ++v) {
        const float2 q = *(const float2*)&s_q[m][mrow][uoff[v]];
        float px = efv * q.x, py = efv * q.y;
        A[m][2 * v]     = (__bf16)px;    // hw v_cvt_pk_bf16_f32
        A[m][2 * v + 1] = (__bf16)py;
      }
#pragma unroll
    for (int nt = 0; nt < 2; ++nt) {
      v16bf Ba = wf[((0 * 2 + nt) * 8 + ks) * 32 + lane];
      v16bf Bb = wf[((1 * 2 + nt) * 8 + ks) * 32 + lane];
      v16bf Bc = wf[((2 * 2 + nt) * 8 + ks) * 32 + lane];
      v16bf Bd = wf[((3 * 2 + nt) * 8 + ks) * 32 + lane];
      accm[0][nt] = wmma_bf16(A[0], Ba, accm[0][nt]);  // Ts  = Zs x WA
      accm[1][nt] = wmma_bf16(A[0], Bb, accm[1][nt]);  // Tb  = Zs x WB
      accm[2][nt] = wmma_bf16(A[1], Bc, accm[2][nt]);  // Tc0 = Z0 x WC
      accm[3][nt] = wmma_bf16(A[2], Bc, accm[3][nt]);  // Tc1 = Z1 x WC
      accm[4][nt] = wmma_bf16(A[3], Bc, accm[4][nt]);  // Tc2 = Z2 x WC
      accm[5][nt] = wmma_bf16(A[4], Bd, accm[5][nt]);  // Td  = Zd x WD
    }
  }

  // per-row (M) scalars for the D-layout (VGPR j -> M = j + hi*8)
  float ysr[8], yvr[8][3];
#pragma unroll
  for (int j = 0; j < 8; ++j) {
    int M = j + hi * 8;
    ysr[j] = s_ys[M];
    yvr[j][0] = s_yv[M][0]; yvr[j][1] = s_yv[M][1]; yvr[j][2] = s_yv[M][2];
  }

  const v16bf* l1s = (const v16bf*)l1sf;
  const v16bf* l1v = (const v16bf*)l1vf;
  // g==0: ms = ys*Ts + Td ; g==1..3: mv_i = yv_i*Tb + ys*Tc_i  -> @L1 -> scatter
#pragma unroll
  for (int g = 0; g < 4; ++g) {
    __syncthreads();
#pragma unroll
    for (int nt = 0; nt < 2; ++nt)
#pragma unroll
      for (int j = 0; j < 8; ++j) {
        float val;
        if (g == 0) val = ysr[j] * accm[0][nt][j] + accm[5][nt][j];
        else        val = yvr[j][g - 1] * accm[1][nt][j] + ysr[j] * accm[1 + g][nt][j];
        s_stage[j + hi * 8][mrow + nt * 16] = val;
      }
    __syncthreads();
    v16bf Af;
#pragma unroll
    for (int v = 0; v < 8; ++v) {
      const float2 q = *(const float2*)&s_stage[mrow][uoff[v]];
      Af[2 * v]     = (__bf16)q.x;
      Af[2 * v + 1] = (__bf16)q.y;
    }
    const v16bf* Bp = (g == 0) ? l1s : l1v;
#pragma unroll
    for (int nt = 0; nt < 2; ++nt) {
      v8f d = wmma_bf16(Af, Bp[nt * 32 + lane], zero);
#pragma unroll
      for (int j = 0; j < 8; ++j) {
        int M = j + hi * 8;
        int n = s_recv[M];
        int N = mrow + nt * 16;
        int col = (g == 0) ? N : (32 + N * 3 + (g - 1));
        atomic_add_f32(&acc[n * 128 + col], d[j]);
      }
    }
  }
}

// ---------------------------------------------------------------------------
// Node kernel: one wave32 per 16-node tile.
// ---------------------------------------------------------------------------
__global__ __launch_bounds__(32)
void node_kernel(const float* __restrict__ node_attrs,
                 const float* __restrict__ acc,
                 const unsigned short* __restrict__ ws0f,
                 const unsigned short* __restrict__ ws1f,
                 const unsigned short* __restrict__ l2sf,
                 const unsigned short* __restrict__ l2vf,
                 float* __restrict__ out) {
  __shared__ float s_M[4][16][32];   // 0:Ms 1..3:Mv_i
  __shared__ float s_attr[16][16];
  __shared__ float s_stage[16][32];

  const int lane = threadIdx.x;
  const int nb = blockIdx.x * 16;

  {
    int e = lane >> 1, h = lane & 1;
    const float* src = acc + (nb + e) * 128 + h * 64;
#pragma unroll
    for (int c = 0; c < 64; ++c) {
      int idx = h * 64 + c;
      float v = src[c];
      if (idx < 32) s_M[0][e][idx] = v;
      else { int z = idx - 32; s_M[1 + (z % 3)][e][z / 3] = v; }
    }
  }
  for (int k = lane; k < 256; k += 32)
    s_attr[k >> 4][k & 15] = node_attrs[(nb + (k >> 4)) * 16 + (k & 15)];
  __syncthreads();

  const int hi = (lane >> 4) & 1;
  const int mrow = lane & 15;
  int koff[8];
#pragma unroll
  for (int v = 0; v < 8; ++v) koff[v] = hi * 16 + 2 * v;

  v8f zero = {0.f, 0.f, 0.f, 0.f, 0.f, 0.f, 0.f, 0.f};
  v8f acc2[4][2];   // 0:ss 1..3:sv_i
#pragma unroll
  for (int g = 0; g < 4; ++g)
#pragma unroll
    for (int nt = 0; nt < 2; ++nt) acc2[g][nt] = zero;

  const v16bf* w0 = (const v16bf*)ws0f;
  const v16bf* w1 = (const v16bf*)ws1f;
  for (int ks = 0; ks < 16; ++ks) {
    v16bf A[4];
#pragma unroll
    for (int m = 0; m < 4; ++m)
#pragma unroll
      for (int v = 0; v < 8; ++v) {
        int K = ks * 32 + koff[v];             // K = u*16 + vv
        int uu = K >> 4, vv = K & 15;          // vv even -> pair {vv, vv+1}
        const float2 a = *(const float2*)&s_attr[mrow][vv];
        float mval = s_M[m][mrow][uu];
        float px = mval * a.x, py = mval * a.y;
        A[m][2 * v]     = (__bf16)px;
        A[m][2 * v + 1] = (__bf16)py;
      }
#pragma unroll
    for (int nt = 0; nt < 2; ++nt) {
      v16bf B0 = w0[(nt * 16 + ks) * 32 + lane];
      v16bf B1 = w1[(nt * 16 + ks) * 32 + lane];
      acc2[0][nt] = wmma_bf16(A[0], B0, acc2[0][nt]);  // ss
      acc2[1][nt] = wmma_bf16(A[1], B1, acc2[1][nt]);  // sv0
      acc2[2][nt] = wmma_bf16(A[2], B1, acc2[2][nt]);  // sv1
      acc2[3][nt] = wmma_bf16(A[3], B1, acc2[3][nt]);  // sv2
    }
  }

  const v16bf* l2s = (const v16bf*)l2sf;
  const v16bf* l2v = (const v16bf*)l2vf;
#pragma unroll
  for (int g = 0; g < 4; ++g) {
    __syncthreads();
#pragma unroll
    for (int nt = 0; nt < 2; ++nt)
#pragma unroll
      for (int j = 0; j < 8; ++j)
        s_stage[j + hi * 8][mrow + nt * 16] = acc2[g][nt][j];
    __syncthreads();
    v16bf Af;
#pragma unroll
    for (int v = 0; v < 8; ++v) {
      const float2 q = *(const float2*)&s_stage[mrow][koff[v]];
      Af[2 * v]     = (__bf16)q.x;
      Af[2 * v + 1] = (__bf16)q.y;
    }
    const v16bf* Bp = (g == 0) ? l2s : l2v;
#pragma unroll
    for (int nt = 0; nt < 2; ++nt) {
      v8f d = wmma_bf16(Af, Bp[nt * 32 + lane], zero);
#pragma unroll
      for (int j = 0; j < 8; ++j) {
        int M = j + hi * 8;
        int N = mrow + nt * 16;
        float base = s_M[g][M][N];             // Ms (g=0) or Mv_{g-1}
        int col = (g == 0) ? N : (32 + N * 3 + (g - 1));
        out[(nb + M) * 128 + col] = base + d[j];
      }
    }
  }
}

extern "C" void kernel_launch(void* const* d_in, const int* in_sizes, int n_in,
                              void* d_out, int out_size, void* d_ws, size_t ws_size,
                              hipStream_t stream) {
  const float* node_attrs = (const float*)d_in[0];
  const float* node_feats = (const float*)d_in[1];
  const float* edge_attrs = (const float*)d_in[2];
  const float* edge_feats = (const float*)d_in[3];
  const int*   edge_index = (const int*)  d_in[4];
  const float* Wgen = (const float*)d_in[5];
  const float* L1s  = (const float*)d_in[6];
  const float* L1v  = (const float*)d_in[7];
  const float* WS0  = (const float*)d_in[8];
  const float* WS1  = (const float*)d_in[9];
  const float* L2s  = (const float*)d_in[10];
  const float* L2v  = (const float*)d_in[11];
  float* out = (float*)d_out;

  char* ws = (char*)d_ws;
  float*          accw  = (float*)(ws + ACC_OFF);
  unsigned short* wfrag = (unsigned short*)(ws + WFRAG_OFF);
  unsigned short* ws0f  = (unsigned short*)(ws + WS0F_OFF);
  unsigned short* ws1f  = (unsigned short*)(ws + WS1F_OFF);
  unsigned short* l1sf  = (unsigned short*)(ws + L1SF_OFF);
  unsigned short* l1vf  = (unsigned short*)(ws + L1VF_OFF);
  unsigned short* l2sf  = (unsigned short*)(ws + L2SF_OFF);
  unsigned short* l2vf  = (unsigned short*)(ws + L2VF_OFF);

  hipLaunchKernelGGL(prep_kernel, dim3(640), dim3(256), 0, stream,
                     Wgen, L1s, L1v, WS0, WS1, L2s, L2v,
                     wfrag, ws0f, ws1f, l1sf, l1vf, l2sf, l2vf, accw);
  hipLaunchKernelGGL(edge_kernel, dim3(EDGE_TILES), dim3(32), 0, stream,
                     node_feats, edge_attrs, edge_feats, edge_index,
                     wfrag, l1sf, l1vf, accw);
  hipLaunchKernelGGL(node_kernel, dim3(NODE_TILES), dim3(32), 0, stream,
                     node_attrs, accw, ws0f, ws1f, l2sf, l2vf, out);
}
